// HypersphericalPrototypeBank_25013889532208
// MI455X (gfx1250) — matrix operand order repaired and checked by reading, not compiled, not run on hardware
//
#include <hip/hip_runtime.h>

typedef __attribute__((ext_vector_type(16))) _Float16 v16h;
typedef __attribute__((ext_vector_type(8)))  _Float16 v8h;
typedef __attribute__((ext_vector_type(8)))  float    v8f;
typedef __attribute__((ext_vector_type(4)))  int      v4i;

#define CCH 512
#define HWPIX 16384           // 128*128
#define NPIX 131072           // 8*16384
#define KCLS 171
#define KPAD 176
#define IGNORE_IDX 255

// workspace layout (in floats)
#define WS_SUMS   0                    // KCLS*CCH floats
#define WS_COUNTS (KCLS*CCH)           // 87552 .. +171
#define WS_LOSS   (WS_COUNTS + KCLS)   // 87723
#define WS_INVT   87728                // 176 floats (32B aligned)
#define WS_PW     87904                // start of half[176*512] B-fragments (32B aligned)
#define PW_HALVES (KPAD*CCH)           // 90112 halves = 180224 bytes

#if defined(__has_builtin)
#if __has_builtin(__builtin_amdgcn_global_load_async_to_lds_b128)
#define HAVE_ASYNC_LDS 1
#endif
#if __has_builtin(__builtin_amdgcn_s_wait_asynccnt)
#define HAVE_WAIT_ASYNC 1
#endif
#endif

typedef __attribute__((address_space(1))) v4i as1_v4i;
typedef __attribute__((address_space(3))) v4i as3_v4i;

// ---------------- kernel 0: zero accumulators ----------------
__global__ void k0_zero(float* ws, int n) {
    int i = blockIdx.x * blockDim.x + threadIdx.x;
    if (i < n) ws[i] = 0.0f;
}

// ---------------- kernel 1: segment sums + counts ----------------
// block = 256 threads = 64 pixels x 4 channel groups
__global__ void __launch_bounds__(256) k1_segsum(const float* __restrict__ features,
                                                 const int* __restrict__ labels,
                                                 float* __restrict__ ws) {
    int t = threadIdx.x;
    int p = blockIdx.x * 64 + (t & 63);
    int cg = t >> 6;
    int lbl = labels[p];
    bool valid = (lbl != IGNORE_IDX) && (lbl >= 0) && (lbl < KCLS);
    if (cg == 0 && valid) atomicAdd(ws + WS_COUNTS + lbl, 1.0f);
    if (!valid) return;
    int b = p >> 14, hw = p & (HWPIX - 1);
    const float* fb = features + (size_t)b * (CCH * HWPIX) + hw;
    float* srow = ws + WS_SUMS + (size_t)lbl * CCH;
    for (int c = cg; c < CCH; c += 4) {
        atomicAdd(srow + c, fb[(size_t)c * HWPIX]);
    }
}

// ---------------- kernel 2: prototype EMA update ----------------
// one block (256 thr) per class; also emits f16 WMMA-B fragments + 1/temp
__global__ void __launch_bounds__(256) k2_protos(const float* __restrict__ protos,
                                                 const float* __restrict__ class_counts,
                                                 const float* __restrict__ temps,
                                                 float* __restrict__ ws,
                                                 float* __restrict__ out_protos,
                                                 float* __restrict__ out_counts) {
    __shared__ float red[256];
    int k = blockIdx.x, t = threadIdx.x;
    _Float16* pw = (_Float16*)(ws + WS_PW);
    int kt = k >> 4, kj = k & 15;
    if (k >= KCLS) {  // padding classes: zero B columns, neutral temp
        for (int c = t; c < CCH; c += 256) {
            int fi = kt * 16 + (c >> 5);
            pw[(size_t)fi * 512 + (size_t)(kj + (((c >> 3) & 1) << 4)) * 16 + (c & 7) + (((c >> 4) & 1) << 3)] = (_Float16)0.0f;
        }
        if (t == 0) ws[WS_INVT + k] = 1.0f;
        return;
    }
    float cnt = ws[WS_COUNTS + k];
    float denom = fmaxf(cnt, 1.0f);
    int c0 = t, c1 = t + 256;
    float m0 = ws[WS_SUMS + (size_t)k * CCH + c0] / denom;
    float m1 = ws[WS_SUMS + (size_t)k * CCH + c1] / denom;
    red[t] = m0 * m0 + m1 * m1;
    __syncthreads();
    for (int s = 128; s > 0; s >>= 1) { if (t < s) red[t] += red[t + s]; __syncthreads(); }
    float invMean = 1.0f / fmaxf(sqrtf(red[0]), 1e-12f);
    __syncthreads();

    float newCount = class_counts[k] + cnt;
    float mm = fminf(0.99f + 0.09f * (1.0f - 1.0f / (1.0f + newCount * 0.001f)), 0.999f);
    float p0 = protos[(size_t)k * CCH + c0], p1 = protos[(size_t)k * CCH + c1];
    float u0 = mm * p0 + (1.0f - mm) * (m0 * invMean);
    float u1 = mm * p1 + (1.0f - mm) * (m1 * invMean);
    red[t] = u0 * u0 + u1 * u1;
    __syncthreads();
    for (int s = 128; s > 0; s >>= 1) { if (t < s) red[t] += red[t + s]; __syncthreads(); }
    float invU = 1.0f / fmaxf(sqrtf(red[0]), 1e-12f);

    bool present = cnt > 0.0f;
    float np0 = present ? u0 * invU : p0;
    float np1 = present ? u1 * invU : p1;
    out_protos[(size_t)k * CCH + c0] = np0;
    out_protos[(size_t)k * CCH + c1] = np1;
    // B-fragment layout: frag fi=(kt*16+cc) holds B[K=channel-in-chunk][N=class-in-tile]
    // lane L owns class N=L%16; half h=L/16, element s holds K=8h+s (s<8) / 16+8h+(s-8) (s>=8).
    int fi0 = kt * 16 + (c0 >> 5);
    pw[(size_t)fi0 * 512 + (size_t)(kj + (((c0 >> 3) & 1) << 4)) * 16 + (c0 & 7) + (((c0 >> 4) & 1) << 3)] = (_Float16)np0;
    int fi1 = kt * 16 + (c1 >> 5);
    pw[(size_t)fi1 * 512 + (size_t)(kj + (((c1 >> 3) & 1) << 4)) * 16 + (c1 & 7) + (((c1 >> 4) & 1) << 3)] = (_Float16)np1;

    if (t == 0) {
        out_counts[k] = newCount;
        float tp = fminf(fmaxf(temps[k], 0.01f), 1.0f);
        ws[WS_INVT + k] = 1.0f / tp;
    }
}

// ---------------- kernel 3: WMMA sim + online log-softmax loss ----------------
// 256 threads = 8 waves; block covers 128 pixels; each wave owns a 16-pixel row tile.
// LDS: 133 KB feature tile + 180 KB prototype B-fragments (async-DMA staged) ~= 315 KB.
__global__ void __launch_bounds__(256) k3_loss(const float* __restrict__ features,
                                               const int* __restrict__ labels,
                                               const float* __restrict__ ws_ro,
                                               float* __restrict__ lossAccum) {
    __shared__ __align__(32) _Float16 fbuf[128 * 520];   // raw f16 features, padded stride
    __shared__ __align__(32) _Float16 pwlds[PW_HALVES];  // prototype B fragments
    __shared__ float ssq[256];
    __shared__ float invn[128];
    __shared__ int   lbls[128];
    const int t = threadIdx.x;
    const int n0 = blockIdx.x * 128;
    const _Float16* pwg = (const _Float16*)(ws_ro + WS_PW);

    // --- phase P: kick off async DMA of the 180 KB prototype table into LDS ---
    // 44 iterations x 256 threads x 16B = 180224 B, tracked on ASYNCcnt.
#ifdef HAVE_ASYNC_LDS
    for (int i = 0; i < 44; ++i) {
        int idx = (i * 256 + t) * 8;  // halves (16B chunks)
        __builtin_amdgcn_global_load_async_to_lds_b128(
            (as1_v4i*)(pwg + idx), (as3_v4i*)(pwlds + idx), 0, 0);
    }
#else
    for (int i = 0; i < 44; ++i) {
        int idx = (i * 256 + t) * 8;
        *(v8h*)(pwlds + idx) = *(const v8h*)(pwg + idx);
    }
#endif

    // --- phase A: load 128 px x 512 ch, f16 to LDS, accumulate sumsq (overlaps DMA) ---
    {
        int p = t & 127, cs = t >> 7;
        int n = n0 + p;
        int b = n >> 14, hw = n & (HWPIX - 1);
        const float* fb = features + (size_t)b * (CCH * HWPIX) + hw;
        float s = 0.0f;
        for (int c = cs; c < CCH; c += 2) {
            float v = fb[(size_t)c * HWPIX];
            s += v * v;
            fbuf[p * 520 + c] = (_Float16)v;
        }
        ssq[t] = s;
    }
    __syncthreads();
    if (t < 128) {
        float tot = ssq[t] + ssq[t + 128];
        invn[t] = 1.0f / fmaxf(sqrtf(tot), 1e-12f);
        lbls[t] = labels[n0 + t];
    }
#ifdef HAVE_ASYNC_LDS
#ifdef HAVE_WAIT_ASYNC
    __builtin_amdgcn_s_wait_asynccnt(0);
#else
    asm volatile("s_wait_asynccnt 0" ::: "memory");
#endif
#endif
    __syncthreads();

    // --- phase B: per-wave 16x176 sim tile via WMMA, streaming softmax ---
    const int lane = t & 31, w = t >> 5;
    const int hi = lane >> 4, li = lane & 15;
    const int prow0 = w * 16;
    const float* invT = ws_ro + WS_INVT;

    float inr[8]; int lb[8];
#pragma unroll
    for (int r = 0; r < 8; ++r) {
        int row = prow0 + r + hi * 8;   // D layout: VGPR r -> M=r (lanes<16) / M=r+8 (lanes>=16)
        inr[r] = invn[row];
        lb[r] = lbls[row];
    }
    float runMax[8], runSum[8], tgtLog[8];
#pragma unroll
    for (int r = 0; r < 8; ++r) { runMax[r] = -3.0e38f; runSum[r] = 0.0f; tgtLog[r] = 0.0f; }

    const int arow = prow0 + li;        // A layout: M = li for both lane halves
    for (int kt = 0; kt < 11; ++kt) {
        v8f acc = {};
#pragma unroll
        for (int cc = 0; cc < 16; ++cc) {
            // A fragment: lanes<16 take K{0..7,16..23}, lanes>=16 take K{8..15,24..31}
            union { v16h v; v8h h[2]; } a;
            const _Float16* ab = &fbuf[arow * 520 + cc * 32 + hi * 8];
            a.h[0] = *(const v8h*)(ab);
            a.h[1] = *(const v8h*)(ab + 16);
            // B fragment: lane L owns class N=L%16, 16 K values of the 32-chunk in A's order
            union { v16h v; v8h h[2]; } b;
            const _Float16* bb = &pwlds[(size_t)(kt * 16 + cc) * 512 + lane * 16];
            b.h[0] = *(const v8h*)(bb);
            b.h[1] = *(const v8h*)(bb + 8);
            acc = __builtin_amdgcn_wmma_f32_16x16x32_f16(false, a.v, false, b.v,
                                                         (short)0, acc, false, false);
        }
        int ck = kt * 16 + li;
        float it = invT[ck];
        bool cvalid = ck < KCLS;
#pragma unroll
        for (int r = 0; r < 8; ++r) {
            float v = acc[r] * inr[r] * it;            // fused normalize + temperature
            if (!cvalid) v = -3.0e38f;                 // mask padded classes
            // target logit: single gather from lane holding class lb[r]
            float tl = __shfl(v, (lb[r] & 15) + (hi << 4), 32);
            if ((lb[r] >> 4) == kt) tgtLog[r] = tl;
            // 16-lane tile max
            float tmax = v;
            tmax = fmaxf(tmax, __shfl_xor(tmax, 1, 32));
            tmax = fmaxf(tmax, __shfl_xor(tmax, 2, 32));
            tmax = fmaxf(tmax, __shfl_xor(tmax, 4, 32));
            tmax = fmaxf(tmax, __shfl_xor(tmax, 8, 32));
            float e = __expf(v - tmax);
            e += __shfl_xor(e, 1, 32);
            e += __shfl_xor(e, 2, 32);
            e += __shfl_xor(e, 4, 32);
            e += __shfl_xor(e, 8, 32);
            // online logsumexp merge
            float nm = fmaxf(runMax[r], tmax);
            runSum[r] = runSum[r] * __expf(runMax[r] - nm) + e * __expf(tmax - nm);
            runMax[r] = nm;
        }
    }
    float total = 0.0f;
#pragma unroll
    for (int r = 0; r < 8; ++r) {
        int l = lb[r];
        bool valid = (l != IGNORE_IDX) && (l >= 0) && (l < KCLS);
        float nll = valid ? (runMax[r] + __logf(runSum[r]) - tgtLog[r]) : 0.0f;
        total += nll;
    }
    if (li == 0) atomicAdd(lossAccum, total);   // lanes 0 and 16: one per half-wave
}

// ---------------- kernel 4: finalize loss ----------------
__global__ void k4_final(const float* __restrict__ ws, float* __restrict__ out) {
    __shared__ float red[256];
    int t = threadIdx.x;
    red[t] = (t < KCLS) ? ws[WS_COUNTS + t] : 0.0f;
    __syncthreads();
    for (int s = 128; s > 0; s >>= 1) { if (t < s) red[t] += red[t + s]; __syncthreads(); }
    if (t == 0) out[0] = ws[WS_LOSS] / fmaxf(red[0], 1.0f);
}

extern "C" void kernel_launch(void* const* d_in, const int* in_sizes, int n_in,
                              void* d_out, int out_size, void* d_ws, size_t ws_size,
                              hipStream_t stream) {
    const float* features = (const float*)d_in[0];
    const float* protos   = (const float*)d_in[1];
    const float* ccounts  = (const float*)d_in[2];
    const float* ctemp    = (const float*)d_in[3];
    const int*   labels   = (const int*)d_in[4];
    float* out = (float*)d_out;
    float* ws  = (float*)d_ws;
    float* out_protos = out + 1;
    float* out_counts = out + 1 + KCLS * CCH;

    const int nz = WS_LOSS + 1;  // sums + counts + loss accumulator
    k0_zero<<<(nz + 255) / 256, 256, 0, stream>>>(ws, nz);
    k1_segsum<<<NPIX / 64, 256, 0, stream>>>(features, labels, ws);
    k2_protos<<<KPAD, 256, 0, stream>>>(protos, ccounts, ctemp, ws, out_protos, out_counts);
    k3_loss<<<NPIX / 128, 256, 0, stream>>>(features, labels, ws, ws + WS_LOSS);
    k4_final<<<1, 256, 0, stream>>>(ws, out);
}
